// TokenRouterGating_86165633893006
// MI455X (gfx1250) — matrix operand order repaired and checked by reading, not compiled, not run on hardware
//
#include <hip/hip_runtime.h>

#define HIDDEN      4096
#define NUM_EXPERTS 64
#define TOPK        8
#define ALPHA_F     0.01f
#define TOKENS      16384
#define KSPLIT      4
#define KRANGE      (HIDDEN / KSPLIT)   // 1024 per wave
#define TOK_PER_BLK 32                  // 2 token tiles * 16; 8 waves (2 tiles x 4 ksplits)
#define LDS_STRIDE  65                  // 64 + 1 pad -> conflict-free column access

typedef __attribute__((ext_vector_type(16))) __bf16 v16bf;
typedef __attribute__((ext_vector_type(4)))  __bf16 v4bf;
typedef __attribute__((ext_vector_type(8)))  float  v8f;
typedef __attribute__((ext_vector_type(4)))  float  v4f;

static __device__ __forceinline__ v4f ld4_nt(const float* p) {
  return __builtin_nontemporal_load((const v4f*)p);
}

static __device__ __forceinline__ v16bf pack16(v4f a, v4f b, v4f c, v4f d) {
  v16bf r;
  r[0]  = (__bf16)a.x; r[1]  = (__bf16)a.y; r[2]  = (__bf16)a.z; r[3]  = (__bf16)a.w;
  r[4]  = (__bf16)b.x; r[5]  = (__bf16)b.y; r[6]  = (__bf16)b.z; r[7]  = (__bf16)b.w;
  r[8]  = (__bf16)c.x; r[9]  = (__bf16)c.y; r[10] = (__bf16)c.z; r[11] = (__bf16)c.w;
  r[12] = (__bf16)d.x; r[13] = (__bf16)d.y; r[14] = (__bf16)d.z; r[15] = (__bf16)d.w;
  return r;
}

// ---- per-launch prep: zero accumulators + convert weights fp32 -> bf16 ----
__global__ void zero_ws_kernel(float* ws) {
  int i = threadIdx.x;
  if (i < 2 * NUM_EXPERTS) ws[i] = 0.0f;
}

__global__ void convert_w_kernel(const float* __restrict__ w, __bf16* __restrict__ wb) {
  int base = (blockIdx.x * blockDim.x + threadIdx.x) * 4;
  if (base < NUM_EXPERTS * HIDDEN) {
    v4f f = *(const v4f*)(w + base);
    v4bf o;
    o[0] = (__bf16)f.x; o[1] = (__bf16)f.y; o[2] = (__bf16)f.z; o[3] = (__bf16)f.w;
    *(v4bf*)(wb + base) = o;
  }
}

__launch_bounds__(256)
__global__ void router_kernel(const float*  __restrict__ x,   // [TOKENS, HIDDEN] fp32
                              const __bf16* __restrict__ wb,  // [NUM_EXPERTS, HIDDEN] bf16
                              float* __restrict__ out_idx,    // [TOKENS, TOPK] (as float)
                              float* __restrict__ out_gate,   // [TOKENS, TOPK]
                              float* __restrict__ ws)         // [0:64) imp sums, [64:128) counts
{
  __shared__ float sP[KSPLIT][TOK_PER_BLK * LDS_STRIDE];  // partial logits per K-split
  __shared__ float s_cnt[NUM_EXPERTS];

  const int tid    = threadIdx.x;
  const int wave   = tid >> 5;          // 0..7
  const int lane   = tid & 31;
  const int half   = lane >> 4;         // which 16-lane half of the wave
  const int lp     = lane & 15;
  const int tile   = wave >> 2;         // 0..1 : which 16-token tile
  const int ksplit = wave & 3;          // 0..3 : which quarter of K

  if (tid < NUM_EXPERTS) s_cnt[tid] = 0.0f;

  const int blockTokenBase = blockIdx.x * TOK_PER_BLK;
  const int waveTokenBase  = blockTokenBase + tile * 16;
  const int kbase          = ksplit * KRANGE;

  // A: lane lp (both halves) streams token row waveTokenBase+lp, different K sub-blocks
  const float* a_row = x + (size_t)(waveTokenBase + lp) * HIDDEN;
  // B: lane lp holds bf16 expert row (nt*16 + lp); halves take K 0-15 / 16-31 of the chunk
  const __bf16* b_row0 = wb + (size_t)(lp)      * HIDDEN;
  const __bf16* b_row1 = wb + (size_t)(16 + lp) * HIDDEN;
  const __bf16* b_row2 = wb + (size_t)(32 + lp) * HIDDEN;
  const __bf16* b_row3 = wb + (size_t)(48 + lp) * HIDDEN;

  v8f acc0 = {}, acc1 = {}, acc2 = {}, acc3 = {};

#pragma unroll 2
  for (int kc = kbase; kc < kbase + KRANGE; kc += 32) {
    // ---- A tile 16x32 bf16: half 0 -> K 0..15; half 1 -> K 16..31
    const int a0 = kc + half * 16;
    const int a1 = kc + half * 16 + 8;
    v16bf A = pack16(ld4_nt(a_row + a0), ld4_nt(a_row + a0 + 4),
                     ld4_nt(a_row + a1), ld4_nt(a_row + a1 + 4));

    // ---- B tiles 32x16 bf16 (preconverted): half 0 -> K 0..15 ; half 1 -> K 16..31
    const int kb = kc + half * 16;
    v16bf B0 = *(const v16bf*)(b_row0 + kb);
    v16bf B1 = *(const v16bf*)(b_row1 + kb);
    v16bf B2 = *(const v16bf*)(b_row2 + kb);
    v16bf B3 = *(const v16bf*)(b_row3 + kb);

    acc0 = __builtin_amdgcn_wmma_f32_16x16x32_bf16(false, A, false, B0, (short)0, acc0, false, false);
    acc1 = __builtin_amdgcn_wmma_f32_16x16x32_bf16(false, A, false, B1, (short)0, acc1, false, false);
    acc2 = __builtin_amdgcn_wmma_f32_16x16x32_bf16(false, A, false, B2, (short)0, acc2, false, false);
    acc3 = __builtin_amdgcn_wmma_f32_16x16x32_bf16(false, A, false, B3, (short)0, acc3, false, false);
  }

  // ---- scatter partial logits: D[m][n], n = lp (+16*nt), m = v + 8*half
  {
    const int mBase = tile * 16 + half * 8;
#pragma unroll
    for (int v = 0; v < 8; ++v) {
      float* p = &sP[ksplit][(mBase + v) * LDS_STRIDE + lp];
      p[0]  = acc0[v];
      p[16] = acc1[v];
      p[32] = acc2[v];
      p[48] = acc3[v];
    }
  }
  __syncthreads();

  // ---- combine the 4 K-split partials into sP[0]
  {
#pragma unroll
    for (int j = 0; j < (TOK_PER_BLK * NUM_EXPERTS) / 256; ++j) {
      const int idx = tid + j * 256;
      const int row = idx >> 6;
      const int e   = idx & 63;
      const int off = row * LDS_STRIDE + e;
      sP[0][off] = sP[0][off] + sP[1][off] + sP[2][off] + sP[3][off];
    }
  }
  __syncthreads();

  // ---- softmax per token (thread t <-> token row t)
  if (tid < TOK_PER_BLK) {
    const int row = tid;
    float m = -1e30f;
    for (int e = 0; e < NUM_EXPERTS; ++e)
      m = fmaxf(m, sP[0][row * LDS_STRIDE + e]);
    float s = 0.0f;
    for (int e = 0; e < NUM_EXPERTS; ++e) {
      float ev = __expf(sP[0][row * LDS_STRIDE + e] - m);
      sP[0][row * LDS_STRIDE + e] = ev;
      s += ev;
    }
    const float invs = 1.0f / s;
    for (int e = 0; e < NUM_EXPERTS; ++e)
      sP[0][row * LDS_STRIDE + e] *= invs;        // now holds probabilities
  }
  __syncthreads();

  // ---- expert importance: column sums of probabilities
  if (tid < NUM_EXPERTS) {
    float cs = 0.0f;
    for (int r = 0; r < TOK_PER_BLK; ++r)
      cs += sP[0][r * LDS_STRIDE + tid];
    atomicAdd(&ws[tid], cs);
  }

  // ---- top-8 per token (non-destructive scans; strict '>' => lowest index on ties)
  if (tid < TOK_PER_BLK) {
    const int row   = tid;
    const int token = blockTokenBase + tid;
    int   idxs[TOPK];
    float pk[TOPK];
    float psum = 0.0f;
#pragma unroll
    for (int k = 0; k < TOPK; ++k) {
      float best = -1.0f;
      int   bi   = 0;
      for (int e = 0; e < NUM_EXPERTS; ++e) {
        float v = sP[0][row * LDS_STRIDE + e];
        bool skip = false;
#pragma unroll
        for (int j = 0; j < k; ++j) skip = skip || (idxs[j] == e);
        if (!skip && v > best) { best = v; bi = e; }
      }
      idxs[k] = bi;
      pk[k]   = best;
      psum   += best;
    }
    const float invg = 1.0f / (psum + 1e-6f);
#pragma unroll
    for (int k = 0; k < TOPK; ++k) {
      out_idx[(size_t)token * TOPK + k]  = (float)idxs[k];
      out_gate[(size_t)token * TOPK + k] = pk[k] * invg;
      atomicAdd(&s_cnt[idxs[k]], 1.0f);
    }
  }
  __syncthreads();

  if (tid < NUM_EXPERTS) atomicAdd(&ws[NUM_EXPERTS + tid], s_cnt[tid]);
}

__global__ void finalize_kernel(const float* __restrict__ ws, float* __restrict__ out_aux) {
  if (threadIdx.x == 0) {
    float s = 0.0f;
    for (int e = 0; e < NUM_EXPERTS; ++e)
      s += (ws[e] / (float)TOKENS) * ws[NUM_EXPERTS + e];
    out_aux[0] = s * ALPHA_F;
  }
}

extern "C" void kernel_launch(void* const* d_in, const int* in_sizes, int n_in,
                              void* d_out, int out_size, void* d_ws, size_t ws_size,
                              hipStream_t stream) {
  const float* x = (const float*)d_in[0];   // hidden_states [4,4096,4096] fp32
  const float* w = (const float*)d_in[1];   // expert_gate_weights [64,4096] fp32

  float* out      = (float*)d_out;
  float* out_idx  = out;                          // [16384,8]
  float* out_gate = out + (size_t)TOKENS * TOPK;  // [16384,8]
  float* out_aux  = out + (size_t)2 * TOKENS * TOPK;

  float*  ws  = (float*)d_ws;                     // 128 floats of accumulators
  __bf16* wbf = (__bf16*)((char*)d_ws + 512);     // bf16 weights, 512 KB

  zero_ws_kernel<<<1, 128, 0, stream>>>(ws);
  convert_w_kernel<<<(NUM_EXPERTS * HIDDEN / 4 + 255) / 256, 256, 0, stream>>>(w, wbf);
  router_kernel<<<TOKENS / TOK_PER_BLK, 256, 0, stream>>>(x, wbf, out_idx, out_gate, ws);
  finalize_kernel<<<1, 64, 0, stream>>>(ws, out_aux);
}
